// GAT_89472758710589
// MI455X (gfx1250) — matrix operand order repaired and checked by reading, not compiled, not run on hardware
//
#include <hip/hip_runtime.h>
#include <hip/hip_bf16.h>
#include <math.h>

#define H1   8
#define C1   8
#define HID  64   // H1*C1
#define OUT  2
#define NEG  0.2f

typedef __attribute__((ext_vector_type(16))) __bf16 v16bf;
typedef __attribute__((ext_vector_type(8)))  float  v8f;

union BFrag { unsigned int u[8]; uint4 q[2]; v16bf v; };

__device__ __forceinline__ unsigned int pack_bf16(float lo, float hi) {
    unsigned int a = __float_as_uint(lo), b = __float_as_uint(hi);
    a = (a + 0x7FFFu + ((a >> 16) & 1u)) >> 16;   // round-to-nearest-even
    b = (b + 0x7FFFu + ((b >> 16) & 1u)) >> 16;
    return (a & 0xFFFFu) | (b << 16);
}

// One-shot f32 -> packed bf16 pair conversion (bandwidth-bound pre-pass).
__global__ void f32_to_bf16pk(const float* __restrict__ src,
                              unsigned int* __restrict__ dst, long npairs) {
    long i = (long)blockIdx.x * blockDim.x + threadIdx.x;
    if (i >= npairs) return;
    float2 p = *(const float2*)(src + 2 * i);
    dst[i] = pack_bf16(p.x, p.y);
}

// ---------------------------------------------------------------------------
// h1[N,64] = x[N,K] @ W1[K,64]  via v_wmma_f32_16x16x32_bf16, bf16 operands
// pre-packed in memory so fragment loads are pure global_load_b128:
//   A frag (16x32, ISA 7.12.2): lane&15 = M row; half selects K base 0/8
//     (+16 for VGPRs 4-7); each dword = 2 consecutive K  -> 2 x b128 per kstep
//   B frag (32x16): lane&15 = N col; same K order as A (dword j = K pair
//     8*half+2(j&3)+16(j>>2)) -> gathered bf16 loads, all L2-resident
//   C/D (16x16 f32): lane&15 = N, VGPR v = M (+8 upper half)
// ---------------------------------------------------------------------------
template <int KSTEPS>
__global__ void gemm1_wmma(const unsigned int* __restrict__ xh,
                           const unsigned int* __restrict__ w1h,
                           float* __restrict__ h1, int nnodes, int ksteps_dyn) {
    int wave   = (int)((blockIdx.x * (unsigned)blockDim.x + threadIdx.x) >> 5);
    int ntiles = (nnodes + 15) >> 4;
    if (wave >= ntiles) return;

    const int ksteps = KSTEPS > 0 ? KSTEPS : ksteps_dyn;
    int lane = threadIdx.x & 31;
    int half = lane >> 4;
    int lm   = lane & 15;
    int r0   = wave << 4;

    v8f acc[4] = {};

    int rowA = r0 + lm;
    if (rowA >= nnodes) rowA = nnodes - 1;            // clamp (wave-uniform tile)
    const unsigned int* xr = xh + (long)rowA * (ksteps * 16);  // row stride: K/2 dwords

    #pragma unroll
    for (int ks = 0; ks < ksteps; ++ks) {
        BFrag A;
        A.q[0] = *(const uint4*)(xr + ks * 16 + half * 4);
        A.q[1] = *(const uint4*)(xr + ks * 16 + half * 4 + 8);
        const unsigned short* wb = (const unsigned short*)w1h;
        #pragma unroll
        for (int nt = 0; nt < 4; ++nt) {
            BFrag B;
            int ncol = nt * 16 + lm;
            #pragma unroll
            for (int j = 0; j < 8; ++j) {
                int kk = ks * 32 + half * 8 + (j & 3) * 2 + (j >> 2) * 16;
                B.u[j] = (unsigned int)wb[(long)kk * HID + ncol] |
                         ((unsigned int)wb[(long)(kk + 1) * HID + ncol] << 16);
            }
            acc[nt] = __builtin_amdgcn_wmma_f32_16x16x32_bf16(
                false, A.v, false, B.v, (short)0, acc[nt], false, false);
        }
    }

    #pragma unroll
    for (int nt = 0; nt < 4; ++nt) {
        #pragma unroll
        for (int v = 0; v < 8; ++v) {
            int row = r0 + v + half * 8;
            if (row < nnodes) h1[(long)row * HID + nt * 16 + lm] = acc[nt][v];
        }
    }
}

// al_src/al_dst[n,h] = sum_c h1[n,h*8+c] * a[h,c]
__global__ void att_logits1(const float* __restrict__ h1,
                            const float* __restrict__ a1s, const float* __restrict__ a1d,
                            float* __restrict__ als, float* __restrict__ ald, long nnodes) {
    long i = (long)blockIdx.x * blockDim.x + threadIdx.x;
    if (i >= nnodes * H1) return;
    long n = i >> 3; int hh = (int)(i & 7);
    const float* hr = h1 + n * HID + hh * C1;
    const float* as = a1s + hh * C1;
    const float* ad = a1d + hh * C1;
    float s = 0.f, d = 0.f;
    #pragma unroll
    for (int c = 0; c < C1; ++c) { float v = hr[c]; s += v * as[c]; d += v * ad[c]; }
    als[i] = s; ald[i] = d;
}

__device__ __forceinline__ void edge_sd(const int* __restrict__ ei, long E, long e,
                                        int& s, int& d) {
    if (e < E) { s = ei[e]; d = ei[E + e]; }
    else       { s = d = (int)(e - E); }          // self-loop
}

// ---- layer-1 softmax passes: one thread per (edge, head) ----
__global__ void l1_max(const int* __restrict__ ei, long E, long ET,
                       const float* __restrict__ als, const float* __restrict__ ald,
                       float* __restrict__ m1) {
    long t = (long)blockIdx.x * blockDim.x + threadIdx.x;
    if (t >= ET * H1) return;
    long e = t >> 3; int hh = (int)(t & 7);
    int s, d; edge_sd(ei, E, e, s, d);
    float v = als[(long)s * H1 + hh] + ald[(long)d * H1 + hh];
    v = v > 0.f ? v : NEG * v;
    atomicMax(&m1[(long)d * H1 + hh], v);
}

__global__ void l1_den(const int* __restrict__ ei, long E, long ET,
                       const float* __restrict__ als, const float* __restrict__ ald,
                       const float* __restrict__ m1, float* __restrict__ den1) {
    long t = (long)blockIdx.x * blockDim.x + threadIdx.x;
    if (t >= ET * H1) return;
    long e = t >> 3; int hh = (int)(t & 7);
    int s, d; edge_sd(ei, E, e, s, d);
    long di = (long)d * H1 + hh;
    float v = als[(long)s * H1 + hh] + ald[di];
    v = v > 0.f ? v : NEG * v;
    atomicAdd(&den1[di], __expf(v - m1[di]));
}

__global__ void l1_agg(const int* __restrict__ ei, long E, long ET,
                       const float* __restrict__ als, const float* __restrict__ ald,
                       const float* __restrict__ m1, const float* __restrict__ den1,
                       const float* __restrict__ h1, float* __restrict__ out1) {
    long t = (long)blockIdx.x * blockDim.x + threadIdx.x;
    if (t >= ET * H1) return;
    long e = t >> 3; int hh = (int)(t & 7);
    int s, d; edge_sd(ei, E, e, s, d);
    long di = (long)d * H1 + hh;
    float v = als[(long)s * H1 + hh] + ald[di];
    v = v > 0.f ? v : NEG * v;
    float alpha = __expf(v - m1[di]) / den1[di];
    const float4* hs = (const float4*)(h1 + (long)s * HID + hh * C1);
    float4 p0 = hs[0], p1 = hs[1];
    float* od = out1 + (long)d * HID + hh * C1;
    atomicAdd(&od[0], alpha * p0.x); atomicAdd(&od[1], alpha * p0.y);
    atomicAdd(&od[2], alpha * p0.z); atomicAdd(&od[3], alpha * p0.w);
    atomicAdd(&od[4], alpha * p1.x); atomicAdd(&od[5], alpha * p1.y);
    atomicAdd(&od[6], alpha * p1.z); atomicAdd(&od[7], alpha * p1.w);
}

// ELU(out1 + b1) -> @W2 -> h2[n,2], layer-2 logits
__global__ void elu_lin2(const float* __restrict__ out1, const float* __restrict__ b1,
                         const float* __restrict__ W2,
                         const float* __restrict__ a2s, const float* __restrict__ a2d,
                         float* __restrict__ h2, float* __restrict__ al2s,
                         float* __restrict__ al2d, long nnodes) {
    long n = (long)blockIdx.x * blockDim.x + threadIdx.x;
    if (n >= nnodes) return;
    const float* r = out1 + n * HID;
    float acc0 = 0.f, acc1 = 0.f;
    #pragma unroll
    for (int k = 0; k < HID; ++k) {
        float v = r[k] + b1[k];
        v = v > 0.f ? v : (__expf(v) - 1.0f);     // ELU
        acc0 += v * W2[k * OUT + 0];
        acc1 += v * W2[k * OUT + 1];
    }
    h2[n * OUT + 0] = acc0;
    h2[n * OUT + 1] = acc1;
    al2s[n] = acc0 * a2s[0] + acc1 * a2s[1];
    al2d[n] = acc0 * a2d[0] + acc1 * a2d[1];
}

// ---- layer-2 passes: one thread per edge ----
__global__ void l2_max(const int* __restrict__ ei, long E, long ET,
                       const float* __restrict__ als, const float* __restrict__ ald,
                       float* __restrict__ m2) {
    long e = (long)blockIdx.x * blockDim.x + threadIdx.x;
    if (e >= ET) return;
    int s, d; edge_sd(ei, E, e, s, d);
    float v = als[s] + ald[d];
    v = v > 0.f ? v : NEG * v;
    atomicMax(&m2[d], v);
}

__global__ void l2_den(const int* __restrict__ ei, long E, long ET,
                       const float* __restrict__ als, const float* __restrict__ ald,
                       const float* __restrict__ m2, float* __restrict__ den2) {
    long e = (long)blockIdx.x * blockDim.x + threadIdx.x;
    if (e >= ET) return;
    int s, d; edge_sd(ei, E, e, s, d);
    float v = als[s] + ald[d];
    v = v > 0.f ? v : NEG * v;
    atomicAdd(&den2[d], __expf(v - m2[d]));
}

__global__ void l2_agg(const int* __restrict__ ei, long E, long ET,
                       const float* __restrict__ als, const float* __restrict__ ald,
                       const float* __restrict__ m2, const float* __restrict__ den2,
                       const float* __restrict__ h2, float* __restrict__ out) {
    long e = (long)blockIdx.x * blockDim.x + threadIdx.x;
    if (e >= ET) return;
    int s, d; edge_sd(ei, E, e, s, d);
    float v = als[s] + ald[d];
    v = v > 0.f ? v : NEG * v;
    float alpha = __expf(v - m2[d]) / den2[d];
    atomicAdd(&out[(long)d * OUT + 0], alpha * h2[(long)s * OUT + 0]);
    atomicAdd(&out[(long)d * OUT + 1], alpha * h2[(long)s * OUT + 1]);
}

__global__ void bias_out(float* __restrict__ out, const float* __restrict__ b2, long n2) {
    long i = (long)blockIdx.x * blockDim.x + threadIdx.x;
    if (i < n2) out[i] += b2[i & 1];
}

__global__ void fill_f32(float* __restrict__ p, long count, float v) {
    long i = (long)blockIdx.x * blockDim.x + threadIdx.x;
    if (i < count) p[i] = v;
}

static inline unsigned cdiv(long a, long b) { return (unsigned)((a + b - 1) / b); }

extern "C" void kernel_launch(void* const* d_in, const int* in_sizes, int n_in,
                              void* d_out, int out_size, void* d_ws, size_t ws_size,
                              hipStream_t stream) {
    const float* x   = (const float*)d_in[0];
    const int*   ei  = (const int*)d_in[1];
    // d_in[2] edge_attr: unused by the reference GATConv
    const float* W1  = (const float*)d_in[3];
    const float* a1s = (const float*)d_in[4];
    const float* a1d = (const float*)d_in[5];
    const float* b1  = (const float*)d_in[6];
    const float* W2  = (const float*)d_in[7];
    const float* a2s = (const float*)d_in[8];
    const float* a2d = (const float*)d_in[9];
    const float* b2  = (const float*)d_in[10];

    long kdim = (long)in_sizes[3] / HID;     // IN = 128
    long N    = (long)in_sizes[0] / kdim;    // 100,000
    long E    = (long)in_sizes[1] / 2;       // 3,200,000
    long ET   = E + N;                       // with self-loops

    // workspace layout (xh/w1h first -> 256B-aligned for b128 fragment loads)
    unsigned int* xh  = (unsigned int*)d_ws;           // N*kdim/2 dwords (packed bf16)
    unsigned int* w1h = xh + N * (kdim / 2);           // kdim*HID/2 dwords
    float* h1   = (float*)(w1h + kdim * (HID / 2));    // N*HID
    float* als1 = h1   + N * HID;     // N*H1
    float* ald1 = als1 + N * H1;      // N*H1
    float* m1   = ald1 + N * H1;      // N*H1
    float* den1 = m1   + N * H1;      // N*H1
    float* out1 = den1 + N * H1;      // N*HID
    float* h2   = out1 + N * HID;     // N*OUT
    float* als2 = h2   + N * OUT;     // N
    float* ald2 = als2 + N;           // N
    float* m2   = ald2 + N;           // N
    float* den2 = m2   + N;           // N
    float* outF = (float*)d_out;      // N*OUT

    const int T = 256;
    const float NEG_INF = -3.0e38f;

    // init accumulators
    fill_f32<<<cdiv(N * H1, T), T, 0, stream>>>(m1,   N * H1, NEG_INF);
    fill_f32<<<cdiv(N * H1, T), T, 0, stream>>>(den1, N * H1, 0.f);
    fill_f32<<<cdiv(N * HID, T), T, 0, stream>>>(out1, N * HID, 0.f);
    fill_f32<<<cdiv(N, T), T, 0, stream>>>(m2,   N, NEG_INF);
    fill_f32<<<cdiv(N, T), T, 0, stream>>>(den2, N, 0.f);
    fill_f32<<<cdiv(N * OUT, T), T, 0, stream>>>(outF, N * OUT, 0.f);

    // bf16 packing pre-pass (x: 51.2MB->25.6MB once; W1 tiny, stays in L2)
    f32_to_bf16pk<<<cdiv(N * kdim / 2, T), T, 0, stream>>>(x,  xh,  N * kdim / 2);
    f32_to_bf16pk<<<cdiv(kdim * HID / 2, T), T, 0, stream>>>(W1, w1h, kdim * HID / 2);

    // layer 1
    long ntiles = (N + 15) / 16;
    unsigned gb = cdiv(ntiles * 32, T);
    if (kdim == 128)
        gemm1_wmma<4><<<gb, T, 0, stream>>>(xh, w1h, h1, (int)N, 4);
    else
        gemm1_wmma<0><<<gb, T, 0, stream>>>(xh, w1h, h1, (int)N, (int)(kdim / 32));
    att_logits1<<<cdiv(N * H1, T), T, 0, stream>>>(h1, a1s, a1d, als1, ald1, N);
    l1_max<<<cdiv(ET * H1, T), T, 0, stream>>>(ei, E, ET, als1, ald1, m1);
    l1_den<<<cdiv(ET * H1, T), T, 0, stream>>>(ei, E, ET, als1, ald1, m1, den1);
    l1_agg<<<cdiv(ET * H1, T), T, 0, stream>>>(ei, E, ET, als1, ald1, m1, den1, h1, out1);

    // ELU + layer-2 linear + logits
    elu_lin2<<<cdiv(N, T), T, 0, stream>>>(out1, b1, W2, a2s, a2d, h2, als2, ald2, N);

    // layer 2
    l2_max<<<cdiv(ET, T), T, 0, stream>>>(ei, E, ET, als2, ald2, m2);
    l2_den<<<cdiv(ET, T), T, 0, stream>>>(ei, E, ET, als2, ald2, m2, den2);
    l2_agg<<<cdiv(ET, T), T, 0, stream>>>(ei, E, ET, als2, ald2, m2, den2, h2, outF);
    bias_out<<<cdiv(N * OUT, T), T, 0, stream>>>(outF, b2, N * OUT);
}
